// SwinTransformerV2CrBlock_84859963834999
// MI455X (gfx1250) — compile-verified, matrix operand order checked by reading, not run on hardware
//
#include <hip/hip_runtime.h>
#include <math.h>

// SwinV2-CR block for MI455X (gfx1250): all GEMMs via v_wmma_f32_16x16x32_f16,
// f16 activation tiles staged into LDS via the Tensor Data Mover (TENSOR_LOAD_TO_LDS),
// weight tiles prefetched with global_prefetch_b8.
// Requires ws_size >= ~403 MB (see layout constants below).

typedef __attribute__((ext_vector_type(16))) _Float16 v16h;
typedef __attribute__((ext_vector_type(8)))  float    v8f;
typedef __attribute__((ext_vector_type(4)))  unsigned int u32x4;
typedef __attribute__((ext_vector_type(8)))  int i32x8;
typedef __attribute__((ext_vector_type(4)))  int i32x4;

#define WMMA16(a, b, c) \
  __builtin_amdgcn_wmma_f32_16x16x32_f16(false, (a), false, (b), (short)0, (c), false, false)

// ---- geometry: B=32, H=W=64, C=384, heads=12, hd=32, win=8x8 (L=64), shift=4, nW=64, Bw=2048
// workspace layout (bytes):
//   [0, 301989888)            : qkv f16  (3 x 2048*12*64*32 halves)   -- later reused as MLP hidden
//   [301989888, 302186496)    : bias f32 (12*64*64)
//   [302186496, 402849792)    : attn out f16 (131072*384)
static constexpr long QKV_HALVES   = 50331648L;      // per q/k/v matrix
static constexpr long BIAS_OFF_B   = 301989888L;
static constexpr long ATTN_OFF_B   = 302186496L;

// window-order token r -> flat global token index (gather == scatter for cyclic roll)
__device__ __forceinline__ int tok_map(int r) {
  int bw = r >> 6, l = r & 63;
  int b  = bw >> 6, wi = bw & 63;
  int gh = ((wi >> 3) << 3) + (l >> 3);
  int gw = ((wi & 7) << 3) + (l & 7);
  int h  = (gh + 4) & 63;
  int w  = (gw + 4) & 63;
  return ((b << 6) + h) * 64 + w;
}

// shifted-window mask region id (9 regions) for window wi, in-window pos l
__device__ __forceinline__ int region_id(int wi, int l) {
  int gh = ((wi >> 3) << 3) + (l >> 3);
  int gw = ((wi & 7) << 3) + (l & 7);
  int rh = (gh < 56) ? 0 : ((gh < 60) ? 1 : 2);
  int rw = (gw < 56) ? 0 : ((gw < 60) ? 1 : 2);
  return rh * 3 + rw;
}

// Load a 16x32 f16 fragment from a row-major [16][ld] tile (ISA 7.12.2 A layout;
// B uses the same loader from an N-major tile). lane r=lane&15 is the M (or N) row,
// K element pairs: VGPR p -> K = 2*(p&3) + 16*(p>>2) + 8*(lane>>4).
__device__ __forceinline__ v16h load_frag(const _Float16* p, int ld) {
  const int lane = threadIdx.x & 31;
  const int r = lane & 15, hi = lane >> 4;
  v16h f;
#pragma unroll
  for (int q = 0; q < 8; ++q) {
    int k = ((q & 3) << 1) + ((q >> 2) << 4) + (hi << 3);
    f[2 * q]     = p[r * ld + k];
    f[2 * q + 1] = p[r * ld + k + 1];
  }
  return f;
}

// TDM: DMA a row-major f16 tile (nrows x rowlen halves, global row stride gstride halves)
// into LDS at byte offset lds_off.  D# per CDNA5 ISA:
//   group0: count=1 | lds_addr | global_addr(57b) | type=2
//   group1: data_size=2B, tensor_dim0=rowlen, tensor_dim1=nrows, tile_dim0=rowlen,
//           tile_dim1=nrows, tensor_dim0_stride=gstride
// Issued per wave (each wave DMAs its own slice and waits on its own TENSORcnt).
__device__ __forceinline__ void tdm_load_f16_tile(
    const _Float16* gsrc, unsigned lds_off,
    unsigned rowlen, unsigned nrows, unsigned gstride) {
  unsigned long long ga = (unsigned long long)(size_t)gsrc;
  u32x4 g0;
  g0[0] = 1u;                                         // count=1 (valid descriptor)
  g0[1] = lds_off;                                    // LDS byte address
  g0[2] = (unsigned)(ga & 0xffffffffu);               // global_addr[31:0]
  g0[3] = (unsigned)((ga >> 32) & 0x1ffffffu) | (2u << 30);  // addr[56:32] | type=2
  i32x8 g1;
  g1[0] = (int)(1u << 16);                            // data_size=1 -> 2 bytes
  g1[1] = (int)((rowlen & 0xffffu) << 16);            // tensor_dim0[15:0]
  g1[2] = (int)(((rowlen >> 16) & 0xffffu) | ((nrows & 0xffffu) << 16)); // td0 hi | td1 lo
  g1[3] = (int)(((nrows >> 16) & 0xffffu) | ((rowlen & 0xffffu) << 16)); // td1 hi | tile_dim0
  g1[4] = (int)(nrows & 0xffffu);                     // tile_dim1 (tile_dim2 = 0)
  g1[5] = (int)gstride;                               // tensor_dim0_stride[31:0]
  g1[6] = 0;                                          // stride hi | dim1_stride lo
  g1[7] = 0;
  i32x4 gz = {0, 0, 0, 0};
#if defined(__clang_major__) && (__clang_major__ >= 23)
  i32x8 gz8 = {0, 0, 0, 0, 0, 0, 0, 0};
  __builtin_amdgcn_tensor_load_to_lds(g0, g1, gz, gz, gz8, 0);
#else
  __builtin_amdgcn_tensor_load_to_lds(g0, g1, gz, gz, 0);
#endif
}

// ---------------------------------------------------------------------------
// 1) relative-position-bias MLP: bias[h][i][j], i,j in [0,64)
// ---------------------------------------------------------------------------
__global__ __launch_bounds__(256) void bias_kernel(
    const float* __restrict__ w1, const float* __restrict__ b1,
    const float* __restrict__ w2, const float* __restrict__ b2,
    float* __restrict__ bias) {
  int idx = blockIdx.x * 256 + threadIdx.x;   // 4096 total
  int i = idx >> 6, j = idx & 63;
  float dh = (float)((i >> 3) - (j >> 3));
  float dw = (float)((i & 7) - (j & 7));
  float r0 = (dh > 0.f ? 1.f : (dh < 0.f ? -1.f : 0.f)) * log1pf(fabsf(dh));
  float r1 = (dw > 0.f ? 1.f : (dw < 0.f ? -1.f : 0.f)) * log1pf(fabsf(dw));
  float acc[12];
#pragma unroll
  for (int h = 0; h < 12; ++h) acc[h] = 0.f;
  for (int t = 0; t < 384; ++t) {
    float hv = fmaxf(r0 * w1[t * 2] + r1 * w1[t * 2 + 1] + b1[t], 0.f);
#pragma unroll
    for (int h = 0; h < 12; ++h) acc[h] += hv * w2[h * 384 + t];
  }
#pragma unroll
  for (int h = 0; h < 12; ++h)
    bias[((long)h * 64 + i) * 64 + j] = acc[h] + b2[h];
}

// ---------------------------------------------------------------------------
// 2) QKV: gathered x (shift+window) @ qkv_w^T + b -> q/k/v f16 [bw][head][l][hd]
//    grid (2048, 18), block 256 (8 waves), tile M=64 N=64 K-step 32
// ---------------------------------------------------------------------------
__global__ __launch_bounds__(256) void qkv_kernel(
    const float* __restrict__ x, const float* __restrict__ w,
    const float* __restrict__ bvec, _Float16* __restrict__ qkv) {
  __shared__ _Float16 As[64][32];
  __shared__ _Float16 Bs[64][32];
  __shared__ int ridx[64];
  const int tid = threadIdx.x;
  const int bm = blockIdx.x, bn = blockIdx.y;
  if (tid < 64) ridx[tid] = tok_map(bm * 64 + tid);
  const int wave = tid >> 5;
  const int mi = wave & 3, nj = wave >> 2;
  v8f acc0 = {}, acc1 = {};
  for (int kc = 0; kc < 384; kc += 32) {
    __syncthreads();
#pragma unroll
    for (int it = 0; it < 2; ++it) {
      int q = it * 256 + tid;           // 0..511 float4 slots (64x32 floats)
      int row = q >> 3, c4 = (q & 7) << 2;
      const float* ap = x + (long)ridx[row] * 384 + kc + c4;
      const float* bp = w + (long)(bn * 64 + row) * 384 + kc + c4;
      float4 va = *(const float4*)ap;
      float4 vb = *(const float4*)bp;
      __builtin_prefetch(ap + 32, 0, 1);   // next K tile -> global_prefetch_b8
      __builtin_prefetch(bp + 32, 0, 1);
      As[row][c4 + 0] = (_Float16)va.x; As[row][c4 + 1] = (_Float16)va.y;
      As[row][c4 + 2] = (_Float16)va.z; As[row][c4 + 3] = (_Float16)va.w;
      Bs[row][c4 + 0] = (_Float16)vb.x; Bs[row][c4 + 1] = (_Float16)vb.y;
      Bs[row][c4 + 2] = (_Float16)vb.z; Bs[row][c4 + 3] = (_Float16)vb.w;
    }
    __syncthreads();
    v16h a  = load_frag(&As[mi * 16][0], 32);
    v16h b0 = load_frag(&Bs[(nj * 2 + 0) * 16][0], 32);
    acc0 = WMMA16(a, b0, acc0);
    v16h b1 = load_frag(&Bs[(nj * 2 + 1) * 16][0], 32);
    acc1 = WMMA16(a, b1, acc1);
  }
  const int lane = tid & 31, hi = lane >> 4, ln = lane & 15;
#pragma unroll
  for (int t = 0; t < 2; ++t) {
    v8f acc = t ? acc1 : acc0;
    int o = bn * 64 + (nj * 2 + t) * 16 + ln;      // 0..1151
    float bb = bvec[o];
    int which = o / 384, ch = o % 384;
    int head = ch >> 5, d = ch & 31;
    _Float16* dst = qkv + (long)which * QKV_HALVES;
#pragma unroll
    for (int j = 0; j < 8; ++j) {
      int m = mi * 16 + j + hi * 8;
      int r = bm * 64 + m;
      int bw = r >> 6, l = r & 63;
      dst[((long)(bw * 12 + head) * 64 + l) * 32 + d] = (_Float16)(acc[j] + bb);
    }
  }
}

// ---------------------------------------------------------------------------
// 3) attention: one block per (window, head); block 128 (4 waves)
// ---------------------------------------------------------------------------
__global__ __launch_bounds__(128) void attn_kernel(
    const _Float16* __restrict__ qkv, const float* __restrict__ bias,
    const float* __restrict__ logit_scale, _Float16* __restrict__ aout) {
  __shared__ _Float16 qs[64][32];
  __shared__ _Float16 ks2[64][32];
  __shared__ _Float16 vt[32][64];       // v transposed: vt[d][l]
  __shared__ float    Ss[64][65];
  __shared__ _Float16 Ps[64][64];
  const int tid = threadIdx.x;
  const int bi = blockIdx.x;
  const int bw = bi / 12, head = bi % 12;
  const long base = (long)(bw * 12 + head) * 64 * 32;
  const _Float16* qp = qkv + base;
  const _Float16* kp = qkv + QKV_HALVES + base;
  const _Float16* vp = qkv + 2 * QKV_HALVES + base;

  if (tid < 64) {                       // normalize q rows
    const _Float16* row = qp + tid * 32;
    float s = 0.f;
#pragma unroll
    for (int d = 0; d < 32; ++d) { float v = (float)row[d]; s += v * v; }
    float inv = 1.f / fmaxf(sqrtf(s), 1e-12f);
#pragma unroll
    for (int d = 0; d < 32; ++d) qs[tid][d] = (_Float16)((float)row[d] * inv);
  } else {                              // normalize k rows
    int t = tid - 64;
    const _Float16* row = kp + t * 32;
    float s = 0.f;
#pragma unroll
    for (int d = 0; d < 32; ++d) { float v = (float)row[d]; s += v * v; }
    float inv = 1.f / fmaxf(sqrtf(s), 1e-12f);
#pragma unroll
    for (int d = 0; d < 32; ++d) ks2[t][d] = (_Float16)((float)row[d] * inv);
  }
#pragma unroll
  for (int it = 0; it < 16; ++it) {     // transpose-copy v
    int q = it * 128 + tid;             // 0..2047
    int l = q >> 5, d = q & 31;
    vt[d][l] = vp[l * 32 + d];
  }
  __syncthreads();

  const float scale = __expf(fminf(logit_scale[head], 4.6051702f));
  const int wave = tid >> 5, lane = tid & 31, hi = lane >> 4, ln = lane & 15;
  const int wi = bw & 63;

  // S = qn @ kn^T  (4 waves x 4 col tiles)
  {
    v16h a = load_frag(&qs[wave * 16][0], 32);
#pragma unroll
    for (int nt = 0; nt < 4; ++nt) {
      v16h b = load_frag(&ks2[nt * 16][0], 32);
      v8f c = {};
      c = WMMA16(a, b, c);
      int n = nt * 16 + ln;
      int idn = region_id(wi, n);
#pragma unroll
      for (int j = 0; j < 8; ++j) {
        int m = wave * 16 + j + hi * 8;
        int idm = region_id(wi, m);
        float v = c[j] * scale + bias[((long)head * 64 + m) * 64 + n] +
                  ((idm != idn) ? -100.f : 0.f);
        Ss[m][n] = v;
      }
    }
  }
  __syncthreads();

  if (tid < 64) {                       // row softmax
    float mx = -1e30f;
#pragma unroll
    for (int n = 0; n < 64; ++n) mx = fmaxf(mx, Ss[tid][n]);
    float sum = 0.f;
#pragma unroll
    for (int n = 0; n < 64; ++n) sum += __expf(Ss[tid][n] - mx);
    float inv = 1.f / sum;
#pragma unroll
    for (int n = 0; n < 64; ++n)
      Ps[tid][n] = (_Float16)(__expf(Ss[tid][n] - mx) * inv);
  }
  __syncthreads();

  // out = P @ v  (4 waves x 2 col tiles, K=64)
#pragma unroll
  for (int nt = 0; nt < 2; ++nt) {
    v8f c = {};
#pragma unroll
    for (int kc = 0; kc < 64; kc += 32) {
      v16h a = load_frag(&Ps[wave * 16][kc], 64);
      v16h b = load_frag(&vt[nt * 16][kc], 64);
      c = WMMA16(a, b, c);
    }
    int d = nt * 16 + ln;
#pragma unroll
    for (int j = 0; j < 8; ++j) {
      int m = wave * 16 + j + hi * 8;
      long r = (long)bw * 64 + m;
      aout[r * 384 + head * 32 + d] = (_Float16)c[j];
    }
  }
}

// ---------------------------------------------------------------------------
// 4) proj + LayerNorm(norm1) + residual (window-reverse + unshift fused)
//    grid 2048, block 256; M=64, N=384 (full row for LN), K=384
//    A tile staged by the Tensor Data Mover (one 8-row DMA per wave).
// ---------------------------------------------------------------------------
__global__ __launch_bounds__(256) void proj_kernel(
    const float* __restrict__ x, const _Float16* __restrict__ attn,
    const float* __restrict__ w, const float* __restrict__ bvec,
    const float* __restrict__ gw, const float* __restrict__ gb,
    float* __restrict__ out) {
  // single LDS arena so TDM lds_addr offsets are well-defined (As at offset 0)
  __shared__ __align__(128) unsigned char smem[30208];
  _Float16 (*As)[32] = (_Float16(*)[32])(smem);                 // 4096 B
  _Float16 (*Bs)[32] = (_Float16(*)[32])(smem + 4096);          // 24576 B
  int*   ridx = (int*)(smem + 28672);                           // 256 B
  float* rsum = (float*)(smem + 28928);
  float* rsq  = rsum + 64;
  float* rmu  = rsum + 128;
  float* rinv = rsum + 192;
  const int tid = threadIdx.x;
  const int bm = blockIdx.x;
  if (tid < 64) { ridx[tid] = tok_map(bm * 64 + tid); rsum[tid] = 0.f; rsq[tid] = 0.f; }
  const int wave = tid >> 5, mi = wave & 3, nh = wave >> 2;
  const int lane = tid & 31, hi = lane >> 4, ln = lane & 15;
  v8f acc[12] = {};
  for (int kc = 0; kc < 384; kc += 32) {
    __syncthreads();
    // A: async TDM, each wave DMAs its 8 rows (8 x 32 halves, stride 384)
    tdm_load_f16_tile(attn + (long)(bm * 64 + wave * 8) * 384 + kc,
                      (unsigned)(wave * 512), 32u, 8u, 384u);
#pragma unroll
    for (int it = 0; it < 12; ++it) {   // B: 3072 float4
      int q = it * 256 + tid;
      int row = q >> 3, c4 = (q & 7) << 2;
      const float* bp = w + (long)row * 384 + kc + c4;
      float4 vb = *(const float4*)bp;
      __builtin_prefetch(bp + 32, 0, 1);
      Bs[row][c4 + 0] = (_Float16)vb.x; Bs[row][c4 + 1] = (_Float16)vb.y;
      Bs[row][c4 + 2] = (_Float16)vb.z; Bs[row][c4 + 3] = (_Float16)vb.w;
    }
    __builtin_amdgcn_s_wait_tensorcnt(0);
    __syncthreads();
    v16h a = load_frag(&As[mi * 16][0], 32);
#pragma unroll
    for (int t = 0; t < 12; ++t) {
      v16h b = load_frag(&Bs[(2 * t + nh) * 16][0], 32);
      acc[t] = WMMA16(a, b, acc[t]);
    }
  }
  // bias + row-moment partials
  float psum[8], psq[8];
#pragma unroll
  for (int j = 0; j < 8; ++j) { psum[j] = 0.f; psq[j] = 0.f; }
#pragma unroll
  for (int t = 0; t < 12; ++t) {
    int o = (2 * t + nh) * 16 + ln;
    float bb = bvec[o];
#pragma unroll
    for (int j = 0; j < 8; ++j) {
      float y = acc[t][j] + bb;
      acc[t][j] = y;
      psum[j] += y; psq[j] += y * y;
    }
  }
#pragma unroll
  for (int j = 0; j < 8; ++j) {
    int m = mi * 16 + j + hi * 8;
    atomicAdd(&rsum[m], psum[j]);
    atomicAdd(&rsq[m], psq[j]);
  }
  __syncthreads();
  if (tid < 64) {
    float mu = rsum[tid] * (1.f / 384.f);
    float var = rsq[tid] * (1.f / 384.f) - mu * mu;
    rmu[tid] = mu; rinv[tid] = rsqrtf(var + 1e-5f);
  }
  __syncthreads();
#pragma unroll
  for (int t = 0; t < 12; ++t) {
    int o = (2 * t + nh) * 16 + ln;
    float gwv = gw[o], gbv = gb[o];
#pragma unroll
    for (int j = 0; j < 8; ++j) {
      int m = mi * 16 + j + hi * 8;
      long tok = ridx[m];
      float lnv = (acc[t][j] - rmu[m]) * rinv[m] * gwv + gbv;
      out[tok * 384 + o] = x[tok * 384 + o] + lnv;
    }
  }
}

// ---------------------------------------------------------------------------
// 5) fc1 + exact GELU -> hidden f16; grid (2048, 24), block 256
// ---------------------------------------------------------------------------
__global__ __launch_bounds__(256) void fc1_kernel(
    const float* __restrict__ x1, const float* __restrict__ w,
    const float* __restrict__ bvec, _Float16* __restrict__ hid) {
  __shared__ _Float16 As[64][32];
  __shared__ _Float16 Bs[64][32];
  const int tid = threadIdx.x;
  const int bm = blockIdx.x, bn = blockIdx.y;
  const int wave = tid >> 5, mi = wave & 3, nj = wave >> 2;
  v8f acc0 = {}, acc1 = {};
  for (int kc = 0; kc < 384; kc += 32) {
    __syncthreads();
#pragma unroll
    for (int it = 0; it < 2; ++it) {
      int q = it * 256 + tid;
      int row = q >> 3, c4 = (q & 7) << 2;
      const float* ap = x1 + (long)(bm * 64 + row) * 384 + kc + c4;
      const float* bp = w + (long)(bn * 64 + row) * 384 + kc + c4;
      float4 va = *(const float4*)ap;
      float4 vb = *(const float4*)bp;
      __builtin_prefetch(ap + 32, 0, 1);
      __builtin_prefetch(bp + 32, 0, 1);
      As[row][c4 + 0] = (_Float16)va.x; As[row][c4 + 1] = (_Float16)va.y;
      As[row][c4 + 2] = (_Float16)va.z; As[row][c4 + 3] = (_Float16)va.w;
      Bs[row][c4 + 0] = (_Float16)vb.x; Bs[row][c4 + 1] = (_Float16)vb.y;
      Bs[row][c4 + 2] = (_Float16)vb.z; Bs[row][c4 + 3] = (_Float16)vb.w;
    }
    __syncthreads();
    v16h a  = load_frag(&As[mi * 16][0], 32);
    v16h b0 = load_frag(&Bs[(nj * 2 + 0) * 16][0], 32);
    acc0 = WMMA16(a, b0, acc0);
    v16h b1 = load_frag(&Bs[(nj * 2 + 1) * 16][0], 32);
    acc1 = WMMA16(a, b1, acc1);
  }
  const int lane = tid & 31, hi = lane >> 4, ln = lane & 15;
#pragma unroll
  for (int t = 0; t < 2; ++t) {
    v8f acc = t ? acc1 : acc0;
    int o = bn * 64 + (nj * 2 + t) * 16 + ln;
    float bb = bvec[o];
#pragma unroll
    for (int j = 0; j < 8; ++j) {
      int m = mi * 16 + j + hi * 8;
      float y = acc[j] + bb;
      float g = 0.5f * y * (1.f + erff(y * 0.70710678f));
      hid[(long)(bm * 64 + m) * 1536 + o] = (_Float16)g;
    }
  }
}

// ---------------------------------------------------------------------------
// 6) fc2 + LayerNorm(norm2) + in-place residual into d_out
//    grid 2048, block 256; M=64, N=384, K=1536; TDM-staged A tile.
// ---------------------------------------------------------------------------
__global__ __launch_bounds__(256) void fc2_kernel(
    const _Float16* __restrict__ hid, const float* __restrict__ w,
    const float* __restrict__ bvec, const float* __restrict__ gw,
    const float* __restrict__ gb, float* __restrict__ out) {
  __shared__ __align__(128) unsigned char smem[29696];
  _Float16 (*As)[32] = (_Float16(*)[32])(smem);                 // 4096 B
  _Float16 (*Bs)[32] = (_Float16(*)[32])(smem + 4096);          // 24576 B
  float* rsum = (float*)(smem + 28672);
  float* rsq  = rsum + 64;
  float* rmu  = rsum + 128;
  float* rinv = rsum + 192;
  const int tid = threadIdx.x;
  const int bm = blockIdx.x;
  if (tid < 64) { rsum[tid] = 0.f; rsq[tid] = 0.f; }
  const int wave = tid >> 5, mi = wave & 3, nh = wave >> 2;
  const int lane = tid & 31, hi = lane >> 4, ln = lane & 15;
  v8f acc[12] = {};
  for (int kc = 0; kc < 1536; kc += 32) {
    __syncthreads();
    tdm_load_f16_tile(hid + (long)(bm * 64 + wave * 8) * 1536 + kc,
                      (unsigned)(wave * 512), 32u, 8u, 1536u);
#pragma unroll
    for (int it = 0; it < 12; ++it) {
      int q = it * 256 + tid;
      int row = q >> 3, c4 = (q & 7) << 2;
      const float* bp = w + (long)row * 1536 + kc + c4;
      float4 vb = *(const float4*)bp;
      __builtin_prefetch(bp + 32, 0, 1);
      Bs[row][c4 + 0] = (_Float16)vb.x; Bs[row][c4 + 1] = (_Float16)vb.y;
      Bs[row][c4 + 2] = (_Float16)vb.z; Bs[row][c4 + 3] = (_Float16)vb.w;
    }
    __builtin_amdgcn_s_wait_tensorcnt(0);
    __syncthreads();
    v16h a = load_frag(&As[mi * 16][0], 32);
#pragma unroll
    for (int t = 0; t < 12; ++t) {
      v16h b = load_frag(&Bs[(2 * t + nh) * 16][0], 32);
      acc[t] = WMMA16(a, b, acc[t]);
    }
  }
  float psum[8], psq[8];
#pragma unroll
  for (int j = 0; j < 8; ++j) { psum[j] = 0.f; psq[j] = 0.f; }
#pragma unroll
  for (int t = 0; t < 12; ++t) {
    int o = (2 * t + nh) * 16 + ln;
    float bb = bvec[o];
#pragma unroll
    for (int j = 0; j < 8; ++j) {
      float y = acc[t][j] + bb;
      acc[t][j] = y;
      psum[j] += y; psq[j] += y * y;
    }
  }
#pragma unroll
  for (int j = 0; j < 8; ++j) {
    int m = mi * 16 + j + hi * 8;
    atomicAdd(&rsum[m], psum[j]);
    atomicAdd(&rsq[m], psq[j]);
  }
  __syncthreads();
  if (tid < 64) {
    float mu = rsum[tid] * (1.f / 384.f);
    float var = rsq[tid] * (1.f / 384.f) - mu * mu;
    rmu[tid] = mu; rinv[tid] = rsqrtf(var + 1e-5f);
  }
  __syncthreads();
#pragma unroll
  for (int t = 0; t < 12; ++t) {
    int o = (2 * t + nh) * 16 + ln;
    float gwv = gw[o], gbv = gb[o];
#pragma unroll
    for (int j = 0; j < 8; ++j) {
      int m = mi * 16 + j + hi * 8;
      long tok = (long)bm * 64 + m;
      float lnv = (acc[t][j] - rmu[m]) * rinv[m] * gwv + gbv;
      float xv = out[tok * 384 + o];          // x1 residual (each element touched once)
      out[tok * 384 + o] = xv + lnv;
    }
  }
}

// ---------------------------------------------------------------------------
extern "C" void kernel_launch(void* const* d_in, const int* in_sizes, int n_in,
                              void* d_out, int out_size, void* d_ws, size_t ws_size,
                              hipStream_t stream) {
  (void)in_sizes; (void)n_in; (void)out_size; (void)ws_size;
  const float* x           = (const float*)d_in[0];
  const float* qkv_w       = (const float*)d_in[1];
  const float* qkv_b       = (const float*)d_in[2];
  const float* proj_w      = (const float*)d_in[3];
  const float* proj_b      = (const float*)d_in[4];
  const float* meta_w1     = (const float*)d_in[5];
  const float* meta_b1     = (const float*)d_in[6];
  const float* meta_w2     = (const float*)d_in[7];
  const float* meta_b2     = (const float*)d_in[8];
  const float* logit_scale = (const float*)d_in[9];
  const float* n1w         = (const float*)d_in[10];
  const float* n1b         = (const float*)d_in[11];
  const float* fc1_w       = (const float*)d_in[12];
  const float* fc1_b       = (const float*)d_in[13];
  const float* fc2_w       = (const float*)d_in[14];
  const float* fc2_b       = (const float*)d_in[15];
  const float* n2w         = (const float*)d_in[16];
  const float* n2b         = (const float*)d_in[17];
  float* out = (float*)d_out;

  char* ws = (char*)d_ws;
  _Float16* qkv_h  = (_Float16*)ws;
  float*    biasws = (float*)(ws + BIAS_OFF_B);
  _Float16* attnws = (_Float16*)(ws + ATTN_OFF_B);
  _Float16* hidws  = (_Float16*)ws;          // reuses qkv region (dead by then)

  bias_kernel<<<16, 256, 0, stream>>>(meta_w1, meta_b1, meta_w2, meta_b2, biasws);
  qkv_kernel <<<dim3(2048, 18), 256, 0, stream>>>(x, qkv_w, qkv_b, qkv_h);
  attn_kernel<<<24576, 128, 0, stream>>>(qkv_h, biasws, logit_scale, attnws);
  proj_kernel<<<2048, 256, 0, stream>>>(x, attnws, proj_w, proj_b, n1w, n1b, out);
  fc1_kernel <<<dim3(2048, 24), 256, 0, stream>>>(out, fc1_w, fc1_b, hidws);
  fc2_kernel <<<2048, 256, 0, stream>>>(hidws, fc2_w, fc2_b, n2w, n2b, out);
}
